// MultiLayerMemoryCachingLSTM_91139206021431
// MI455X (gfx1250) — matrix operand order repaired and checked by reading, not compiled, not run on hardware
//
#include <hip/hip_runtime.h>
#include <stdint.h>

#define Bb 16
#define Tt 256
#define Dd 1024
#define Hh 1024
#define Ss 16
#define SCALE_ 0.03125f  // 1/sqrt(1024)

typedef __attribute__((ext_vector_type(16))) __bf16 v16bf;
typedef __attribute__((ext_vector_type(8)))  float  v8f;
typedef __attribute__((ext_vector_type(8)))  unsigned short v8us;
typedef __attribute__((ext_vector_type(16))) unsigned short v16us;
typedef unsigned short bfbits;

#define ACT_NONE 0
#define ACT_RELU 1
#define ACT_SIG  2
#define ACT_ZERO 3

__device__ __forceinline__ bfbits f2bf(float f) {
  unsigned u = __builtin_bit_cast(unsigned, f);
  u += 0x7FFFu + ((u >> 16) & 1u);          // round-to-nearest-even
  return (bfbits)(u >> 16);
}

__device__ __forceinline__ float sigm(float x) { return 1.f / (1.f + __expf(-x)); }

__device__ __forceinline__ v8f wmma_bf16(v16bf a, v16bf b, v8f c) {
  return __builtin_amdgcn_wmma_f32_16x16x32_bf16(false, a, false, b, (short)0, c,
                                                 false, false);
}

// A fragment (16x32 bf16), ISA 16-bit A layout:
// lane (m = lane&15, khi = lane>>4) needs k = khi*8 + {0..7} and 16+khi*8+{0..7}
// of row m -> exactly two 16-byte contiguous runs => 2 x global_load_b128.
__device__ __forceinline__ v16bf load_a_bf(const bfbits* A, int lda, int kbase) {
  const int lane = threadIdx.x & 31;
  const int m = lane & 15, khi = lane >> 4;
  const bfbits* row = A + (long)m * lda + kbase + khi * 8;
  v8us lo = *(const v8us*)(const void*)row;
  v8us hi = *(const v8us*)(const void*)(row + 16);
  v16us t = __builtin_shufflevector(lo, hi, 0, 1, 2, 3, 4, 5, 6, 7,
                                    8, 9, 10, 11, 12, 13, 14, 15);
  return __builtin_bit_cast(v16bf, t);
}

// M=16 GEMM: C[16xN] = act( [A1|A2][16xK] * Bp + bias ).
// Bp is pre-packed per (nt,kt) tile: 512 bf16, lane-major (lane*16 + e),
// lane holds column n=lane&15, k_local = 8*(lane>>4) + e (e<8), 16 + 8*(lane>>4) + (e-8) (e>=8).
__device__ void gemm16(const bfbits* A1, const bfbits* A2, int K1, int K,
                       const bfbits* Bp, int N, const float* bias,
                       float* C, int ldc, int act,
                       bfbits* Cbf, int ldcb,
                       int wave, int nwaves) {
  const int lane = threadIdx.x & 31;
  const int KT = K >> 5, NT = N >> 4;
  for (int nt = wave; nt < NT; nt += nwaves) {
    v8f c = {};
    const bfbits* bp = Bp + (((long)nt * KT) << 9) + (lane << 4);
#pragma unroll 8
    for (int kt = 0; kt < KT; ++kt) {
      const int kb = kt << 5;
      v16bf a = (kb < K1) ? load_a_bf(A1, K1, kb)
                          : load_a_bf(A2, K - K1, kb - K1);
      v16bf b = *(const v16bf*)(const void*)bp;
      __builtin_prefetch((const void*)(bp + 8 * 512), 0, 0);
      bp += 512;
      c = wmma_bf16(a, b, c);
    }
    const int n = (nt << 4) | (lane & 15);
    const float bb = bias ? bias[n] : 0.f;
#pragma unroll
    for (int v = 0; v < 8; ++v) {
      const int m = v + ((lane >> 4) << 3);
      float val = c[v] + bb;
      if (act == ACT_RELU) val = fmaxf(val, 0.f);
      else if (act == ACT_SIG) val = sigm(val);
      else if (act == ACT_ZERO) val = 0.f;
      C[(long)m * ldc + n] = val;
      if (Cbf) Cbf[(long)m * ldcb + n] = f2bf(val);
    }
  }
}

__device__ __forceinline__ void gsync(unsigned* bar, unsigned& lgen) {
  __syncthreads();
  if (threadIdx.x == 0) {
    const unsigned target = lgen + 1;
    __threadfence();
    if (atomicAdd(&bar[0], 1u) == gridDim.x - 1) {
      bar[0] = 0;
      __threadfence();
      atomicAdd(&bar[1], 1u);
    } else {
      volatile unsigned* gen = &bar[1];
      while ((int)(*gen - target) < 0) __builtin_amdgcn_s_sleep(2);
    }
    __threadfence();
    lgen = target;
  }
  __syncthreads();
}

// ---- pre-kernels -----------------------------------------------------------

// Pack W (rows n, K1(+K2) columns from two matrices) transposed into WMMA
// B-fragment tile order. B(k,n) = W[n][k] since all GEMMs are x @ W.T.
__global__ void pack_b2(const float* __restrict__ W1, const float* __restrict__ W2,
                        bfbits* __restrict__ dst, int N, int K1, int K2) {
  const int K = K1 + K2;
  const int KT = K >> 5;
  const long total = (long)N * K;
  for (long i = (long)blockIdx.x * blockDim.x + threadIdx.x; i < total;
       i += (long)gridDim.x * blockDim.x) {
    const long tile = i >> 9;
    const int e = (int)(i & 511);
    const int lane = e >> 4, ve = e & 15;
    const int nt = (int)(tile / KT), kt = (int)(tile % KT);
    const int kl = ((ve >> 3) << 4) | ((lane >> 4) << 3) | (ve & 7);
    const int n = (nt << 4) | (lane & 15);
    const int k = (kt << 5) + kl;
    const float w = (k < K1) ? W1[(long)n * K1 + k] : W2[(long)n * K2 + (k - K1)];
    dst[i] = f2bf(w);
  }
}

__global__ void bias_sum(const float* __restrict__ a, const float* __restrict__ b,
                         float* __restrict__ o, int n) {
  const int i = blockIdx.x * blockDim.x + threadIdx.x;
  if (i < n) o[i] = a[i] + b[i];
}

// One-time f32 -> bf16 cast of x (keeps [B,T,D] layout).
__global__ void cast_bf16(const float* __restrict__ X, bfbits* __restrict__ Y,
                          long n) {
  for (long i = (long)blockIdx.x * blockDim.x + threadIdx.x; i < n;
       i += (long)gridDim.x * blockDim.x)
    Y[i] = f2bf(X[i]);
}

// Input projection: xp_pre[t][b][:] = x[b][t][:] @ W_ip.T + b_ip
// A rows come straight from the bf16 copy of x: row m=b at xbf[(b*Tt+t)*Dd].
__global__ void __launch_bounds__(256)
input_gemm(const bfbits* __restrict__ xbf, const bfbits* __restrict__ Pip,
           const float* __restrict__ bip, float* __restrict__ xp_pre) {
  const int lane = threadIdx.x & 31;
  const int wave = (int)((blockIdx.x * blockDim.x + threadIdx.x) >> 5);
  const int nw = (int)(((long)gridDim.x * blockDim.x) >> 5);
  const int NT = Dd >> 4, KT = Dd >> 5;
  for (int tile = wave; tile < Tt * NT; tile += nw) {
    const int t = tile / NT, nt = tile % NT;
    v8f c = {};
    const bfbits* bp = Pip + (((long)nt * KT) << 9) + (lane << 4);
    const bfbits* Abase = xbf + (long)t * Dd;
#pragma unroll 8
    for (int kt = 0; kt < KT; ++kt) {
      v16bf a = load_a_bf(Abase, Tt * Dd, kt << 5);
      v16bf b = *(const v16bf*)(const void*)bp;
      bp += 512;
      c = wmma_bf16(a, b, c);
    }
    const int n = (nt << 4) | (lane & 15);
    const float bb = bip[n];
#pragma unroll
    for (int v = 0; v < 8; ++v) {
      const int m = v + ((lane >> 4) << 3);
      xp_pre[((long)t * Bb + m) * Dd + n] = c[v] + bb;
    }
  }
}

__global__ void __launch_bounds__(256)
ln_relu_bf16(const float* __restrict__ X, const float* __restrict__ g,
             const float* __restrict__ be, bfbits* __restrict__ Y, int rows) {
  const int lane = threadIdx.x & 31;
  const int wave = (int)((blockIdx.x * blockDim.x + threadIdx.x) >> 5);
  const int nw = (int)(((long)gridDim.x * blockDim.x) >> 5);
  for (int r = wave; r < rows; r += nw) {
    const float* row = X + (long)r * Dd;
    float s1 = 0.f, s2 = 0.f;
    for (int k = lane; k < Dd; k += 32) { float v = row[k]; s1 += v; s2 += v * v; }
    for (int off = 16; off; off >>= 1) { s1 += __shfl_xor(s1, off); s2 += __shfl_xor(s2, off); }
    const float mean = s1 * (1.f / Dd);
    const float var = s2 * (1.f / Dd) - mean * mean;
    const float inv = rsqrtf(var + 1e-5f);
    for (int k = lane; k < Dd; k += 32) {
      const float v = (row[k] - mean) * inv * g[k] + be[k];
      Y[(long)r * Dd + k] = f2bf(fmaxf(v, 0.f));
    }
  }
}

// ---- persistent scan kernel ------------------------------------------------

struct ScanP {
  const bfbits *xp, *P0, *P1, *Pq, *Pk, *Pv, *Pg1, *Pg2, *Po, *Pf;
  const float *bsum0, *bsum1, *bg1, *bg2, *bo, *bfv, *gf, *bef;
  float *z, *h0, *c0, *h1, *c1, *segsum, *q, *retr, *gate, *tmpf, *enh, *scores;
  float *kbank, *vbank;
  bfbits *h0b, *h1b, *gib, *g1b, *mixb, *cfb, *segmb;
  unsigned *bar;
  float *out;
};

__global__ void __launch_bounds__(256) scan_kernel(ScanP P) {
  const int tid = blockIdx.x * blockDim.x + threadIdx.x;
  const int nth = gridDim.x * blockDim.x;
  const int wave = tid >> 5;
  const int nwaves = nth >> 5;
  const int lane = threadIdx.x & 31;
  unsigned lgen = 0;

  for (int t = 0; t < Tt; ++t) {
    const int seg = t >> 4;  // #cached segments
    const bfbits* xt = P.xp + (long)t * (Bb * Dd);

    // z0 = [x_t | h0] @ [W_ih0 | W_hh0]^T + (b_ih0 + b_hh0)
    gemm16(xt, P.h0b, Dd, 2 * Dd, P.P0, 4 * Hh, P.bsum0, P.z, 4 * Hh,
           ACT_NONE, nullptr, 0, wave, nwaves);
    gsync(P.bar, lgen);

    for (int i = tid; i < Bb * Hh; i += nth) {  // LSTM cell 0
      const int b = i >> 10, h = i & 1023;
      const float zi = P.z[b * 4096 + h],        zf = P.z[b * 4096 + 1024 + h];
      const float zg = P.z[b * 4096 + 2048 + h], zo = P.z[b * 4096 + 3072 + h];
      const float c2 = sigm(zf) * P.c0[i] + sigm(zi) * tanhf(zg);
      const float hv = sigm(zo) * tanhf(c2);
      P.c0[i] = c2; P.h0[i] = hv; P.h0b[i] = f2bf(hv);
    }
    gsync(P.bar, lgen);

    gemm16(P.h0b, P.h1b, Hh, 2 * Hh, P.P1, 4 * Hh, P.bsum1, P.z, 4 * Hh,
           ACT_NONE, nullptr, 0, wave, nwaves);
    gsync(P.bar, lgen);

    for (int i = tid; i < Bb * Hh; i += nth) {  // LSTM cell 1 + segsum + cf[:, :H]
      const int b = i >> 10, h = i & 1023;
      const float zi = P.z[b * 4096 + h],        zf = P.z[b * 4096 + 1024 + h];
      const float zg = P.z[b * 4096 + 2048 + h], zo = P.z[b * 4096 + 3072 + h];
      const float c2 = sigm(zf) * P.c1[i] + sigm(zi) * tanhf(zg);
      const float hv = sigm(zo) * tanhf(c2);
      P.c1[i] = c2; P.h1[i] = hv;
      const bfbits hb = f2bf(hv);
      P.h1b[i] = hb;
      P.cfb[b * 2048 + h] = hb;
      P.segsum[i] += hv;
    }
    gsync(P.bar, lgen);

    gemm16(P.h1b, nullptr, Hh, Hh, P.Pq, Hh, nullptr, P.q, Hh,
           ACT_NONE, nullptr, 0, wave, nwaves);
    gsync(P.bar, lgen);

    for (int pr = wave; pr < Bb * Ss; pr += nwaves) {  // scores: wave per (b,s)
      const int b = pr >> 4, s = pr & 15;
      float acc = 0.f;
      const float* qb = P.q + (long)b * Hh;
      const float* kb = P.kbank + ((long)s * Bb + b) * Hh;
      for (int k = lane; k < Hh; k += 32) acc += qb[k] * kb[k];
      for (int off = 16; off; off >>= 1) acc += __shfl_xor(acc, off);
      if (lane == 0) P.scores[b * Ss + s] = (s < seg) ? acc * SCALE_ : -1e9f;
    }
    gsync(P.bar, lgen);

    if (tid < Bb) {  // softmax over s
      float mx = -3.0e38f;
      for (int s = 0; s < Ss; ++s) mx = fmaxf(mx, P.scores[tid * Ss + s]);
      float sum = 0.f;
      for (int s = 0; s < Ss; ++s) {
        const float e = __expf(P.scores[tid * Ss + s] - mx);
        P.scores[tid * Ss + s] = e; sum += e;
      }
      const float inv = 1.f / sum;
      for (int s = 0; s < Ss; ++s) P.scores[tid * Ss + s] *= inv;
    }
    gsync(P.bar, lgen);

    for (int i = tid; i < Bb * Hh; i += nth) {  // retrieve + gate input
      const int b = i >> 10, h = i & 1023;
      float r = 0.f;
#pragma unroll 4
      for (int s = 0; s < Ss; ++s)
        r += P.scores[b * Ss + s] * P.vbank[((long)s * Bb + b) * Hh + h];
      P.retr[i] = r;
      const float h1v = P.h1[i];
      P.gib[b * 3072 + h]        = f2bf(h1v);
      P.gib[b * 3072 + 1024 + h] = f2bf(r);
      P.gib[b * 3072 + 2048 + h] = f2bf(h1v - r);
    }
    gsync(P.bar, lgen);

    gemm16(P.gib, nullptr, 3 * Hh, 3 * Hh, P.Pg1, Hh, P.bg1, P.tmpf, Hh,
           ACT_RELU, P.g1b, Hh, wave, nwaves);
    gsync(P.bar, lgen);

    gemm16(P.g1b, nullptr, Hh, Hh, P.Pg2, Hh, P.bg2, P.gate, Hh,
           ACT_SIG, nullptr, 0, wave, nwaves);
    gsync(P.bar, lgen);

    for (int i = tid; i < Bb * Hh; i += nth) {  // gated mix
      const float g = P.gate[i];
      P.mixb[i] = f2bf(g * P.h1[i] + (1.f - g) * P.retr[i]);
    }
    gsync(P.bar, lgen);

    // out = mix @ Wo.T + bo; ret = (seg>0)? out : 0; write cf[:, H:] directly
    gemm16(P.mixb, nullptr, Hh, Hh, P.Po, Hh, P.bo, P.tmpf, Hh,
           (seg > 0) ? ACT_NONE : ACT_ZERO, P.cfb + Hh, 2 * Hh, wave, nwaves);
    gsync(P.bar, lgen);

    gemm16(P.cfb, nullptr, 2 * Hh, 2 * Hh, P.Pf, Hh, P.bfv, P.enh, Hh,
           ACT_NONE, nullptr, 0, wave, nwaves);
    gsync(P.bar, lgen);

    for (int r = wave; r < Bb; r += nwaves) {  // LN + ReLU -> out[b][t][:]
      const float* row = P.enh + (long)r * Hh;
      float s1 = 0.f, s2 = 0.f;
      for (int k = lane; k < Hh; k += 32) { const float v = row[k]; s1 += v; s2 += v * v; }
      for (int off = 16; off; off >>= 1) { s1 += __shfl_xor(s1, off); s2 += __shfl_xor(s2, off); }
      const float mean = s1 * (1.f / Hh);
      const float var = s2 * (1.f / Hh) - mean * mean;
      const float inv = rsqrtf(var + 1e-5f);
      float* orow = P.out + (long)r * Tt * Hh + (long)t * Hh;
      for (int k = lane; k < Hh; k += 32) {
        const float v = (row[k] - mean) * inv * P.gf[k] + P.bef[k];
        orow[k] = fmaxf(v, 0.f);
      }
    }

    if ((t & 15) == 15) {  // segment end: bank update
      for (int i = tid; i < Bb * Hh; i += nth) {
        P.segmb[i] = f2bf(P.segsum[i] * (1.f / 16.f));
        P.segsum[i] = 0.f;
      }
      gsync(P.bar, lgen);
      gemm16(P.segmb, nullptr, Hh, Hh, P.Pk, Hh, nullptr,
             P.kbank + (long)seg * Bb * Hh, Hh, ACT_NONE, nullptr, 0, wave, nwaves);
      gemm16(P.h1b, nullptr, Hh, Hh, P.Pv, Hh, nullptr,
             P.vbank + (long)seg * Bb * Hh, Hh, ACT_NONE, nullptr, 0, wave, nwaves);
    }
    gsync(P.bar, lgen);
  }
}

// ---- host ------------------------------------------------------------------

extern "C" void kernel_launch(void* const* d_in, const int* in_sizes, int n_in,
                              void* d_out, int out_size, void* d_ws, size_t ws_size,
                              hipStream_t stream) {
  (void)in_sizes; (void)n_in; (void)out_size; (void)ws_size;
  const float* x     = (const float*)d_in[0];
  const float* W_ip  = (const float*)d_in[1];
  const float* b_ip  = (const float*)d_in[2];
  const float* g_ip  = (const float*)d_in[3];
  const float* be_ip = (const float*)d_in[4];
  const float* W_ih0 = (const float*)d_in[5];
  const float* W_hh0 = (const float*)d_in[6];
  const float* b_ih0 = (const float*)d_in[7];
  const float* b_hh0 = (const float*)d_in[8];
  const float* W_ih1 = (const float*)d_in[9];
  const float* W_hh1 = (const float*)d_in[10];
  const float* b_ih1 = (const float*)d_in[11];
  const float* b_hh1 = (const float*)d_in[12];
  const float* Wq    = (const float*)d_in[13];
  const float* Wk    = (const float*)d_in[14];
  const float* Wv    = (const float*)d_in[15];
  const float* Wg1   = (const float*)d_in[16];
  const float* bg1   = (const float*)d_in[17];
  const float* Wg2   = (const float*)d_in[18];
  const float* bg2   = (const float*)d_in[19];
  const float* Wo    = (const float*)d_in[20];
  const float* bo    = (const float*)d_in[21];
  const float* Wf    = (const float*)d_in[22];
  const float* bfv   = (const float*)d_in[23];
  const float* g_f   = (const float*)d_in[24];
  const float* be_f  = (const float*)d_in[25];
  float* out = (float*)d_out;

  char* base = (char*)d_ws;
  size_t off = 0;
  auto alloc = [&](size_t bytes) -> void* {
    void* r = base + off;
    off = (off + bytes + 255) & ~(size_t)255;
    return r;
  };

  // state region (zeroed each launch)
  float* h0     = (float*)alloc((size_t)Bb * Hh * 4);
  float* c0     = (float*)alloc((size_t)Bb * Hh * 4);
  float* h1     = (float*)alloc((size_t)Bb * Hh * 4);
  float* c1     = (float*)alloc((size_t)Bb * Hh * 4);
  float* segsum = (float*)alloc((size_t)Bb * Hh * 4);
  float* kbank  = (float*)alloc((size_t)Ss * Bb * Hh * 4);
  float* vbank  = (float*)alloc((size_t)Ss * Bb * Hh * 4);
  unsigned* bar = (unsigned*)alloc(256);
  const size_t stateBytes = off;

  // packed bf16 weights (L2-resident working set ~56 MB)
  bfbits* P0  = (bfbits*)alloc((size_t)2048 * 4096 * 2);
  bfbits* P1  = (bfbits*)alloc((size_t)2048 * 4096 * 2);
  bfbits* Pq  = (bfbits*)alloc((size_t)1024 * 1024 * 2);
  bfbits* Pk  = (bfbits*)alloc((size_t)1024 * 1024 * 2);
  bfbits* Pv  = (bfbits*)alloc((size_t)1024 * 1024 * 2);
  bfbits* Pg1 = (bfbits*)alloc((size_t)3072 * 1024 * 2);
  bfbits* Pg2 = (bfbits*)alloc((size_t)1024 * 1024 * 2);
  bfbits* Po  = (bfbits*)alloc((size_t)1024 * 1024 * 2);
  bfbits* Pf  = (bfbits*)alloc((size_t)2048 * 1024 * 2);
  bfbits* Pip = (bfbits*)alloc((size_t)1024 * 1024 * 2);

  bfbits* xbf    = (bfbits*)alloc((size_t)Bb * Tt * Dd * 2);
  bfbits* xp     = (bfbits*)alloc((size_t)Tt * Bb * Dd * 2);
  float*  xp_pre = (float*)alloc((size_t)Tt * Bb * Dd * 4);
  float*  z      = (float*)alloc((size_t)Bb * 4096 * 4);
  float*  bsum0  = (float*)alloc((size_t)4096 * 4);
  float*  bsum1  = (float*)alloc((size_t)4096 * 4);
  float*  q      = (float*)alloc((size_t)Bb * Hh * 4);
  float*  retr   = (float*)alloc((size_t)Bb * Hh * 4);
  float*  gate   = (float*)alloc((size_t)Bb * Hh * 4);
  float*  tmpf   = (float*)alloc((size_t)Bb * Hh * 4);
  float*  enh    = (float*)alloc((size_t)Bb * Hh * 4);
  float*  scores = (float*)alloc((size_t)Bb * Ss * 4);
  bfbits* h0b   = (bfbits*)alloc((size_t)Bb * Hh * 2);
  bfbits* h1b   = (bfbits*)alloc((size_t)Bb * Hh * 2);
  bfbits* gib   = (bfbits*)alloc((size_t)Bb * 3 * Hh * 2);
  bfbits* g1b   = (bfbits*)alloc((size_t)Bb * Hh * 2);
  bfbits* mixb  = (bfbits*)alloc((size_t)Bb * Hh * 2);
  bfbits* cfb   = (bfbits*)alloc((size_t)Bb * 2 * Hh * 2);
  bfbits* segmb = (bfbits*)alloc((size_t)Bb * Hh * 2);

  hipMemsetAsync(d_ws, 0, stateBytes, stream);

  const dim3 tb(256);
  pack_b2<<<4096, tb, 0, stream>>>(W_ih0, W_hh0, P0, 4096, 1024, 1024);
  pack_b2<<<4096, tb, 0, stream>>>(W_ih1, W_hh1, P1, 4096, 1024, 1024);
  pack_b2<<<1024, tb, 0, stream>>>(Wq,  Wq,  Pq,  1024, 1024, 0);
  pack_b2<<<1024, tb, 0, stream>>>(Wk,  Wk,  Pk,  1024, 1024, 0);
  pack_b2<<<1024, tb, 0, stream>>>(Wv,  Wv,  Pv,  1024, 1024, 0);
  pack_b2<<<2048, tb, 0, stream>>>(Wg1, Wg1, Pg1, 1024, 3072, 0);
  pack_b2<<<1024, tb, 0, stream>>>(Wg2, Wg2, Pg2, 1024, 1024, 0);
  pack_b2<<<1024, tb, 0, stream>>>(Wo,  Wo,  Po,  1024, 1024, 0);
  pack_b2<<<2048, tb, 0, stream>>>(Wf,  Wf,  Pf,  1024, 2048, 0);
  pack_b2<<<1024, tb, 0, stream>>>(W_ip, W_ip, Pip, 1024, 1024, 0);
  bias_sum<<<16, tb, 0, stream>>>(b_ih0, b_hh0, bsum0, 4096);
  bias_sum<<<16, tb, 0, stream>>>(b_ih1, b_hh1, bsum1, 4096);
  cast_bf16<<<2048, tb, 0, stream>>>(x, xbf, (long)Bb * Tt * Dd);

  input_gemm<<<2048, tb, 0, stream>>>(xbf, Pip, b_ip, xp_pre);
  ln_relu_bf16<<<512, tb, 0, stream>>>(xp_pre, g_ip, be_ip, xp, Tt * Bb);

  ScanP P;
  P.xp = xp; P.P0 = P0; P.P1 = P1; P.Pq = Pq; P.Pk = Pk; P.Pv = Pv;
  P.Pg1 = Pg1; P.Pg2 = Pg2; P.Po = Po; P.Pf = Pf;
  P.bsum0 = bsum0; P.bsum1 = bsum1; P.bg1 = bg1; P.bg2 = bg2; P.bo = bo;
  P.bfv = bfv; P.gf = g_f; P.bef = be_f;
  P.z = z; P.h0 = h0; P.c0 = c0; P.h1 = h1; P.c1 = c1; P.segsum = segsum;
  P.q = q; P.retr = retr; P.gate = gate; P.tmpf = tmpf; P.enh = enh;
  P.scores = scores; P.kbank = kbank; P.vbank = vbank;
  P.h0b = h0b; P.h1b = h1b; P.gib = gib; P.g1b = g1b; P.mixb = mixb;
  P.cfb = cfb; P.segmb = segmb;
  P.bar = bar; P.out = out;

  scan_kernel<<<128, tb, 0, stream>>>(P);
}
